// NQueensRecurrentRelationalNet_85203561218568
// MI455X (gfx1250) — matrix operand order repaired and checked by reading, not compiled, not run on hardware
//
#include <hip/hip_runtime.h>

// ---------------------------------------------------------------------------
// Types for CDNA5 WMMA (wave32): v_wmma_f32_16x16x32_bf16
// ---------------------------------------------------------------------------
typedef __attribute__((ext_vector_type(16))) __bf16 v16bf;
typedef __attribute__((ext_vector_type(8)))  float  v8f;
typedef __attribute__((ext_vector_type(4)))  unsigned int u32x4;

union FragBF {
    v16bf v;
    u32x4 q[2];
};

#define HDIM 96
#define MAXDEG 16

// ---------------------------------------------------------------------------
// Generic WMMA GEMM:  out = act( A1[idx1] @ W1  (+ A2[idx2] @ W2)  + bias )
//   PARTS (1|2): compile-time -> branch-free two-part (concat) GEMM.
//   MT (M-tiles per wave = 2): each B fragment loaded from LDS feeds
//   MT (x PARTS) back-to-back WMMAs -> halves ds traffic per WMMA and gives
//   the scheduler independent WMMA work behind each wait.
//   A matrices: bf16 row-major, row length K (same K both parts).
//   W matrices: fp32 row-major [K, ldw]; block handles a 96-wide N chunk at
//   blockIdx.y*96; weights staged transposed (bf16) into LDS.
//   8 waves/block; each wave = MT*16 rows x 6 N-tiles (96 cols).
// ---------------------------------------------------------------------------
template <int PARTS, int MT>
__global__ void __launch_bounds__(256)
gemm_bf16_wmma(const __bf16* __restrict__ A1, const int* __restrict__ idx1,
               const __bf16* __restrict__ A2, const int* __restrict__ idx2,
               int idxStride,
               const float* __restrict__ W1, const float* __restrict__ W2,
               const float* __restrict__ bias,
               int M, int K, int ldw, int Nout, int act,
               __bf16* __restrict__ outB, float* __restrict__ outF)
{
    extern __shared__ __bf16 sm[];      // [(PARTS*96) cols][K] transposed weights
    const int ncBase = blockIdx.y * 96;

    // Stage weights fp32 -> bf16, transposed: sm[(p*96+col)*K + k] = Wp[k, ncBase+col]
    {
        const int total = PARTS * 96 * K;
        for (int i = threadIdx.x; i < total; i += 256) {
            int p    = i / (96 * K);
            int rem  = i - p * 96 * K;
            int colL = rem / K;
            int k    = rem - colL * K;
            const float* Wp = (PARTS == 2 && p) ? W2 : W1;
            sm[i] = (__bf16)Wp[(size_t)k * ldw + ncBase + colL];
        }
    }
    __syncthreads();

    const int lane = threadIdx.x & 31;
    const int wave = threadIdx.x >> 5;
    const int half = lane >> 4;          // K-half selector
    const int ml   = lane & 15;          // row (A/C) or col (B/C) within tile
    const int row0 = (blockIdx.x * 8 + wave) * (16 * MT);
    if (row0 >= M) return;               // uniform per-wave guard (after barrier)

    const __bf16* a1p[MT];
    const __bf16* a2p[MT];
    #pragma unroll
    for (int mt = 0; mt < MT; ++mt) {
        const int rowM = row0 + mt * 16 + ml;
        long r1 = rowM, r2 = rowM;
        if (idx1) r1 = idx1[(long)rowM * idxStride];
        if (PARTS == 2 && idx2) r2 = idx2[(long)rowM * idxStride];
        a1p[mt] = A1 + (long)r1 * K;
        a2p[mt] = (PARTS == 2) ? (A2 + (long)r2 * K) : nullptr;
    }

    v8f acc[MT][6] = {};   // MT x 6 N-tiles x (16x16 f32 / 32 lanes = 8 VGPRs)

    for (int kb = 0; kb < K; kb += 32) {
        // A fragments per ISA 16-bit A 16x32 layout:
        //   elements 0..7  -> K = kb + half*8 + (0..7)
        //   elements 8..15 -> K = kb + 16 + half*8 + (0..7)
        FragBF a1[MT], a2[MT];
        #pragma unroll
        for (int mt = 0; mt < MT; ++mt) {
            a1[mt].q[0] = *reinterpret_cast<const u32x4*>(a1p[mt] + kb + half * 8);
            a1[mt].q[1] = *reinterpret_cast<const u32x4*>(a1p[mt] + kb + 16 + half * 8);
            if constexpr (PARTS == 2) {
                a2[mt].q[0] = *reinterpret_cast<const u32x4*>(a2p[mt] + kb + half * 8);
                a2[mt].q[1] = *reinterpret_cast<const u32x4*>(a2p[mt] + kb + 16 + half * 8);
            }
        }

        #pragma unroll
        for (int nt = 0; nt < 6; ++nt) {
            const int colL = nt * 16 + ml;
            // B fragment: elements 0..7 -> K = kb + half*8 + (0..7), 8..15 -> K = kb + 16 + half*8 + (0..7)
            FragBF b1;
            const __bf16* bp = sm + (size_t)colL * K + kb + half * 8;
            b1.q[0] = *reinterpret_cast<const u32x4*>(bp);
            b1.q[1] = *reinterpret_cast<const u32x4*>(bp + 16);
            FragBF b2;
            if constexpr (PARTS == 2) {
                const __bf16* bp2 = sm + (size_t)(96 + colL) * K + kb + half * 8;
                b2.q[0] = *reinterpret_cast<const u32x4*>(bp2);
                b2.q[1] = *reinterpret_cast<const u32x4*>(bp2 + 16);
            }
            // Each B fragment feeds MT (x PARTS) independent WMMAs.
            #pragma unroll
            for (int mt = 0; mt < MT; ++mt) {
                acc[mt][nt] = __builtin_amdgcn_wmma_f32_16x16x32_bf16(
                    false, a1[mt].v, false, b1.v, (short)0, acc[mt][nt], false, false);
            }
            if constexpr (PARTS == 2) {
                #pragma unroll
                for (int mt = 0; mt < MT; ++mt) {
                    acc[mt][nt] = __builtin_amdgcn_wmma_f32_16x16x32_bf16(
                        false, a2[mt].v, false, b2.v, (short)0, acc[mt][nt], false, false);
                }
            }
        }
    }

    // Epilogue: C/D layout -> lane holds N = ml, rows M = half*8 + e
    #pragma unroll
    for (int mt = 0; mt < MT; ++mt) {
        const int mbase = row0 + mt * 16 + half * 8;
        #pragma unroll
        for (int nt = 0; nt < 6; ++nt) {
            const int colG = ncBase + nt * 16 + ml;
            const float bv = bias[colG];
            #pragma unroll
            for (int e = 0; e < 8; ++e) {
                float v = acc[mt][nt][e] + bv;
                if (act) v = fmaxf(v, 0.0f);
                const size_t off = (size_t)(mbase + e) * Nout + colG;
                if (outB) outB[off] = (__bf16)v;
                if (outF) outF[off] = v;
            }
        }
    }
}

// ---------------------------------------------------------------------------
// Build node input: xin = concat(emb[quiz], rows, cols)  (bf16, [NN, 384])
// ---------------------------------------------------------------------------
extern "C" __global__ void prep_kernel(const int* __restrict__ quiz,
                                       const float* __restrict__ rows,
                                       const float* __restrict__ cols,
                                       const float* __restrict__ emb,
                                       __bf16* __restrict__ xin, int NN)
{
    const long i = (long)blockIdx.x * blockDim.x + threadIdx.x;
    const long total = (long)NN * 384;
    if (i >= total) return;
    const int n = (int)(i / 384);
    const int f = (int)(i - (long)n * 384);
    float v;
    if (f < 128)       v = emb[quiz[n] * 128 + f];
    else if (f < 256)  v = rows[(long)n * 128 + (f - 128)];
    else               v = cols[(long)n * 128 + (f - 256)];
    xin[i] = (__bf16)v;
}

// ---------------------------------------------------------------------------
// Zero LSTM state
// ---------------------------------------------------------------------------
extern "C" __global__ void zero_state(float* __restrict__ c, float* __restrict__ hf,
                                      __bf16* __restrict__ hb, int total)
{
    const long i = (long)blockIdx.x * blockDim.x + threadIdx.x;
    if (i >= total) return;
    c[i] = 0.0f; hf[i] = 0.0f; hb[i] = (__bf16)0.0f;
}

// ---------------------------------------------------------------------------
// Static graph: per-node incoming edge table (deterministic, no atomics)
// ---------------------------------------------------------------------------
extern "C" __global__ void build_inedges(const int* __restrict__ ei, int epb,
                                         int* __restrict__ inlist,
                                         int* __restrict__ indeg, int NN)
{
    const int n = blockIdx.x * blockDim.x + threadIdx.x;
    if (n >= NN) return;
    const int b = n >> 4;                // board id (16 cells/board)
    const int base = b * epb;
    int cnt = 0;
    for (int e = 0; e < epb; ++e) {
        const int dst = ei[(long)(base + e) * 2 + 1];
        if (dst == n && cnt < MAXDEG) inlist[n * MAXDEG + cnt++] = base + e;
    }
    indeg[n] = cnt;
}

// ---------------------------------------------------------------------------
// Segment sum over incoming edges: agg[n,f] = sum_e msg[e,f]   (fp32 accum)
// ---------------------------------------------------------------------------
extern "C" __global__ void aggregate_kernel(const __bf16* __restrict__ msg,
                                            const int* __restrict__ inlist,
                                            const int* __restrict__ indeg,
                                            __bf16* __restrict__ agg, int NN)
{
    const long i = (long)blockIdx.x * blockDim.x + threadIdx.x;
    const long total = (long)NN * HDIM;
    if (i >= total) return;
    const int n = (int)(i / HDIM);
    const int f = (int)(i - (long)n * HDIM);
    const int deg = indeg[n];
    float s = 0.0f;
    for (int d = 0; d < deg; ++d) {
        const long e = inlist[n * MAXDEG + d];
        s += (float)msg[e * HDIM + f];
    }
    agg[i] = (__bf16)s;
}

// ---------------------------------------------------------------------------
// LSTM elementwise: gates from z = [i|j|f|o] (fp32, [NN,384])
// ---------------------------------------------------------------------------
__device__ __forceinline__ float sigmoidf_(float x) { return 1.0f / (1.0f + expf(-x)); }

extern "C" __global__ void lstm_update(const float* __restrict__ z,
                                       float* __restrict__ c,
                                       float* __restrict__ hf,
                                       __bf16* __restrict__ hb, int NN)
{
    const long i = (long)blockIdx.x * blockDim.x + threadIdx.x;
    const long total = (long)NN * HDIM;
    if (i >= total) return;
    const int n = (int)(i / HDIM);
    const int j = (int)(i - (long)n * HDIM);
    const float zi = z[(long)n * 384 + j];
    const float zj = z[(long)n * 384 + 96 + j];
    const float zf = z[(long)n * 384 + 192 + j];
    const float zo = z[(long)n * 384 + 288 + j];
    float cv = c[i];
    cv = cv * sigmoidf_(zf + 1.0f) + sigmoidf_(zi) * tanhf(zj);
    const float hv = tanhf(cv) * sigmoidf_(zo);
    c[i]  = cv;
    hf[i] = hv;
    hb[i] = (__bf16)hv;
}

// ---------------------------------------------------------------------------
// Output projection: out[n, 0:5] = h[n] @ out_W + out_b  (N=5 too small for WMMA)
// ---------------------------------------------------------------------------
extern "C" __global__ void out_proj(const float* __restrict__ hf,
                                    const float* __restrict__ W,
                                    const float* __restrict__ b,
                                    float* __restrict__ out, int NN)
{
    const int n = blockIdx.x * blockDim.x + threadIdx.x;
    if (n >= NN) return;
    float acc[5];
    #pragma unroll
    for (int cc = 0; cc < 5; ++cc) acc[cc] = b[cc];
    for (int k = 0; k < HDIM; ++k) {
        const float hv = hf[(long)n * HDIM + k];
        #pragma unroll
        for (int cc = 0; cc < 5; ++cc) acc[cc] += hv * W[k * 5 + cc];
    }
    #pragma unroll
    for (int cc = 0; cc < 5; ++cc) out[(long)n * 5 + cc] = acc[cc];
}

// ---------------------------------------------------------------------------
// Host launcher
// ---------------------------------------------------------------------------
struct NetParams {
    const float *emb;
    const float *preW[4],  *preB[4];
    const float *msgW[4],  *msgB[4];
    const float *postW[4], *postB[4];
    const float *lstmW, *lstmB, *outW, *outB;
};

extern "C" void kernel_launch(void* const* d_in, const int* in_sizes, int n_in,
                              void* d_out, int out_size, void* d_ws, size_t ws_size,
                              hipStream_t stream)
{
    const int*   quiz = (const int*)d_in[0];
    const float* rows = (const float*)d_in[1];
    const float* cols = (const float*)d_in[2];
    const int*   ei   = (const int*)d_in[3];

    const int NN  = in_sizes[0];          // 16384 nodes
    const int E   = in_sizes[3] / 2;      // 155648 edges
    const int epb = E / (NN / 16);        // 152 edges per board

    NetParams p{};
    if (n_in > 5) {
        // Params as individual leaves (order-agnostic: anchor by unique sizes;
        // within each MLP the leaves are contiguous W,b,W,b,... after its W1).
        for (int i = 4; i < n_in; ++i) {
            const int s = in_sizes[i];
            const float* q = (const float*)d_in[i];
            if (s == 640)        p.emb = q;
            else if (s == 36864) { for (int l = 0; l < 4; ++l) { p.preW[l]  = (const float*)d_in[i + 2*l]; p.preB[l]  = (const float*)d_in[i + 2*l + 1]; } }
            else if (s == 18528) { for (int l = 0; l < 4; ++l) { p.msgW[l]  = (const float*)d_in[i + 2*l]; p.msgB[l]  = (const float*)d_in[i + 2*l + 1]; } }
            else if (s == 18432) { for (int l = 0; l < 4; ++l) { p.postW[l] = (const float*)d_in[i + 2*l]; p.postB[l] = (const float*)d_in[i + 2*l + 1]; } }
            else if (s == 73728) p.lstmW = q;
            else if (s == 384)   p.lstmB = q;
            else if (s == 480)   p.outW  = q;
            else if (s == 5)     p.outB  = q;
        }
    } else {
        // Fallback: params passed as a single flat blob in dict-insertion order.
        const float* base = (const float*)d_in[4];
        size_t o = 0;
        p.emb = base + o; o += 640;
        for (int l = 0; l < 4; ++l) { p.preW[l]  = base + o; o += (l == 0 ? 384 : 96) * 96; p.preB[l]  = base + o; o += 96; }
        for (int l = 0; l < 4; ++l) { p.msgW[l]  = base + o; o += (l == 0 ? 193 : 96) * 96; p.msgB[l]  = base + o; o += 96; }
        for (int l = 0; l < 4; ++l) { p.postW[l] = base + o; o += (l == 0 ? 192 : 96) * 96; p.postB[l] = base + o; o += 96; }
        p.lstmW = base + o; o += 192 * 384;
        p.lstmB = base + o; o += 384;
        p.outW  = base + o; o += 96 * 5;
        p.outB  = base + o; o += 5;
    }

    // ------------------------- workspace carve-out -------------------------
    size_t off = 0;
    auto alloc = [&](size_t bytes) -> char* {
        char* q = (char*)d_ws + off;
        off += (bytes + 255) & ~(size_t)255;
        return q;
    };
    __bf16* xin  = (__bf16*)alloc((size_t)NN * 384 * 2);
    __bf16* x0b  = (__bf16*)alloc((size_t)NN * 96 * 2);
    __bf16* hb   = (__bf16*)alloc((size_t)NN * 96 * 2);
    __bf16* nb0  = (__bf16*)alloc((size_t)NN * 96 * 2);
    __bf16* nb1  = (__bf16*)alloc((size_t)NN * 96 * 2);
    __bf16* aggb = (__bf16*)alloc((size_t)NN * 96 * 2);
    float*  hf   = (float*)alloc((size_t)NN * 96 * 4);
    float*  cst  = (float*)alloc((size_t)NN * 96 * 4);
    float*  zb   = (float*)alloc((size_t)NN * 384 * 4);
    __bf16* eb0  = (__bf16*)alloc((size_t)E * 96 * 2);
    __bf16* eb1  = (__bf16*)alloc((size_t)E * 96 * 2);
    int* inlist  = (int*)alloc((size_t)NN * MAXDEG * 4);
    int* indeg   = (int*)alloc((size_t)NN * 4);
    (void)ws_size;

    const int* srcIdx = ei;       // stride-2 columns of edge_index
    const int* dstIdx = ei + 1;

    constexpr int MT = 2;
    const int rowsPerBlock = 8 * 16 * MT;   // 256

    // Single-part GEMM
    auto gemm1 = [&](const __bf16* A1, const int* i1,
                     const float* W1, const float* bias,
                     int M, int K, int ldw, int Nout, int act,
                     __bf16* ob, float* of) {
        dim3 grid((M + rowsPerBlock - 1) / rowsPerBlock, Nout / 96);
        size_t sh = (size_t)96 * K * 2;
        gemm_bf16_wmma<1, MT><<<grid, 256, sh, stream>>>(A1, i1, nullptr, nullptr, 0,
                                                         W1, nullptr, bias, M, K, ldw,
                                                         Nout, act, ob, of);
    };
    // Two-part (concat) GEMM
    auto gemm2 = [&](const __bf16* A1, const int* i1, const __bf16* A2, const int* i2,
                     int istride, const float* W1, const float* W2, const float* bias,
                     int M, int K, int ldw, int Nout, int act,
                     __bf16* ob, float* of) {
        dim3 grid((M + rowsPerBlock - 1) / rowsPerBlock, Nout / 96);
        size_t sh = (size_t)2 * 96 * K * 2;
        gemm_bf16_wmma<2, MT><<<grid, 256, sh, stream>>>(A1, i1, A2, i2, istride,
                                                         W1, W2, bias, M, K, ldw,
                                                         Nout, act, ob, of);
    };

    // ----------------------------- pre stage -------------------------------
    {
        long t = (long)NN * 384;
        prep_kernel<<<(unsigned)((t + 255) / 256), 256, 0, stream>>>(quiz, rows, cols, p.emb, xin, NN);
    }
    gemm1(xin, nullptr, p.preW[0], p.preB[0], NN, 384, 96, 96, 1, nb0, nullptr);
    gemm1(nb0, nullptr, p.preW[1], p.preB[1], NN,  96, 96, 96, 1, nb1, nullptr);
    gemm1(nb1, nullptr, p.preW[2], p.preB[2], NN,  96, 96, 96, 1, nb0, nullptr);
    gemm1(nb0, nullptr, p.preW[3], p.preB[3], NN,  96, 96, 96, 0, x0b, nullptr);

    {
        long t = (long)NN * 96;
        zero_state<<<(unsigned)((t + 255) / 256), 256, 0, stream>>>(cst, hf, hb, (int)t);
    }
    build_inedges<<<(NN + 255) / 256, 256, 0, stream>>>(ei, epb, inlist, indeg, NN);

    // --------------------------- recurrent steps ---------------------------
    float* outf = (float*)d_out;
    for (int t = 0; t < 32; ++t) {
        const __bf16* xcur = (t == 0) ? x0b : hb;

        // msg MLP (ef column is identically zero -> drop weight row 192)
        gemm2(xcur, srcIdx, xcur, dstIdx, 2,
              p.msgW[0], p.msgW[0] + 96 * 96, p.msgB[0], E, 96, 96, 96, 1, eb1, nullptr);
        gemm1(eb1, nullptr, p.msgW[1], p.msgB[1], E, 96, 96, 96, 1, eb0, nullptr);
        gemm1(eb0, nullptr, p.msgW[2], p.msgB[2], E, 96, 96, 96, 1, eb1, nullptr);
        gemm1(eb1, nullptr, p.msgW[3], p.msgB[3], E, 96, 96, 96, 0, eb0, nullptr);

        // deterministic segment-sum over dst
        {
            long tt = (long)NN * 96;
            aggregate_kernel<<<(unsigned)((tt + 255) / 256), 256, 0, stream>>>(eb0, inlist, indeg, aggb, NN);
        }

        // post MLP on concat(agg, x0)
        gemm2(aggb, nullptr, x0b, nullptr, 0,
              p.postW[0], p.postW[0] + 96 * 96, p.postB[0], NN, 96, 96, 96, 1, nb0, nullptr);
        gemm1(nb0, nullptr, p.postW[1], p.postB[1], NN, 96, 96, 96, 1, nb1, nullptr);
        gemm1(nb1, nullptr, p.postW[2], p.postB[2], NN, 96, 96, 96, 1, nb0, nullptr);
        gemm1(nb0, nullptr, p.postW[3], p.postB[3], NN, 96, 96, 96, 0, nb1, nullptr);

        // LSTM: z = concat(y, h) @ lstm_W + b   (N = 384, fp32 out)
        gemm2(nb1, nullptr, hb, nullptr, 0,
              p.lstmW, p.lstmW + 96 * 384, p.lstmB, NN, 96, 384, 384, 0, nullptr, zb);
        {
            long tt = (long)NN * 96;
            lstm_update<<<(unsigned)((tt + 255) / 256), 256, 0, stream>>>(zb, cst, hf, hb, NN);
        }

        // per-step output
        out_proj<<<(NN + 255) / 256, 256, 0, stream>>>(hf, p.outW, p.outB, outf + (size_t)t * NN * 5, NN);
    }
    (void)out_size;
}
